// GameTensor_27195732918735
// MI455X (gfx1250) — compile-verified, matrix operand order checked by reading, not run on hardware
//
#include <hip/hip_runtime.h>
#include <math.h>

#define NAG   4
#define BATCH 256
#define DLAT  128
#define HID   256   // 2*D
#define NPAIR 12    // ordered pairs i != j

typedef __attribute__((ext_vector_type(16))) _Float16 v16h;
typedef __attribute__((ext_vector_type(8)))  float    v8f;

union Frag16 {
    v16h v;
    _Float16 h[16];
    uint4 q[2];
};

// A-matrix (16x32 f16) lane layout per CDNA5 ISA 7.12.2:
// lane L holds row M = L&15; element e of the 16-half fragment maps to
// K' = (e&7) + ((e>>3)<<4) + ((L>>4)<<3)
// (L<16: K in {0..7,16..23}; L>=16: K in {8..15,24..31}).
// B (32x16) uses the symmetric layout with N = L&15, same K' map, so the
// un-scaled fragments of A=W1^T and B=W1 are the SAME buffer.

// ---------------------------------------------------------------------------
// Kernel 0: swizzle W1 (f32, [N,256,128]) into f16 WMMA-fragment order:
// swz[agent][kb(8)][tile(8)][lane(32)][e(16)]
// ---------------------------------------------------------------------------
__global__ void swizzle_w1_kernel(const float* __restrict__ W1,
                                  _Float16* __restrict__ swz) {
    int tid  = blockIdx.x * blockDim.x + threadIdx.x;   // 0..8191
    int lane = tid & 31;
    int tile = (tid >> 5) & 7;
    int kb   = (tid >> 8) & 7;
    int ag   = tid >> 11;
    if (ag >= NAG) return;
    int half = lane >> 4;
    int m    = tile * 16 + (lane & 15);
    _Float16* dst = swz + ((size_t)(((ag * 8 + kb) * 8 + tile) * 32 + lane)) * 16;
#pragma unroll
    for (int e = 0; e < 16; ++e) {
        int kp = (e & 7) + ((e >> 3) << 4) + (half << 3);
        int k  = kb * 32 + kp;
        dst[e] = (_Float16)W1[((size_t)(ag * HID + k)) * DLAT + m];
    }
}

// ---------------------------------------------------------------------------
// Kernel 1: c[pair][b][k] = W2_k * (-2 t (1 - t^2)),  t = tanh(W1[i] z[j,b] + b1[i])
// One block per (b, pair); thread k handles hidden unit k.
// ---------------------------------------------------------------------------
__global__ void coeff_kernel(const float* __restrict__ z_all,
                             const float* __restrict__ W1,
                             const float* __restrict__ b1,
                             const float* __restrict__ W2,
                             float* __restrict__ cbuf) {
    __shared__ float z_sh[DLAT];
    int b    = blockIdx.x;
    int pair = blockIdx.y;
    int i    = pair / 3;
    int jr   = pair % 3;
    int j    = jr + (jr >= i ? 1 : 0);
    int k    = threadIdx.x;  // 0..255

    if (k < DLAT) z_sh[k] = z_all[((size_t)(j * BATCH + b)) * DLAT + k];
    __syncthreads();

    const float4* wrow = (const float4*)(W1 + ((size_t)(i * HID + k)) * DLAT);
    float a = b1[i * HID + k];
#pragma unroll 8
    for (int d = 0; d < DLAT / 4; ++d) {
        float4 w = wrow[d];
        a = fmaf(w.x, z_sh[4 * d + 0], a);
        a = fmaf(w.y, z_sh[4 * d + 1], a);
        a = fmaf(w.z, z_sh[4 * d + 2], a);
        a = fmaf(w.w, z_sh[4 * d + 3], a);
    }
    float t = tanhf(a);
    float c = W2[i * HID + k] * (-2.0f * t * (1.0f - t * t));
    cbuf[((size_t)(pair * BATCH + b)) * HID + k] = c;
}

// ---------------------------------------------------------------------------
// Kernel 2: zero the i==j diagonal blocks of the output [N,N,B,D,D].
// ---------------------------------------------------------------------------
__global__ void zero_diag_kernel(float* __restrict__ out) {
    const size_t per_block_f4 = (size_t)BATCH * DLAT * DLAT / 4;  // 1,048,576
    const size_t total = (size_t)NAG * per_block_f4;
    float4 z4 = make_float4(0.f, 0.f, 0.f, 0.f);
    for (size_t idx = (size_t)blockIdx.x * blockDim.x + threadIdx.x; idx < total;
         idx += (size_t)gridDim.x * blockDim.x) {
        size_t i = idx / per_block_f4;
        size_t w = idx % per_block_f4;
        ((float4*)out)[(i * NAG + i) * per_block_f4 + w] = z4;
    }
}

// ---------------------------------------------------------------------------
// Kernel 3: H[i,j,b] = W1[i]^T diag(c) W1[i]  via f16 WMMA, f32 accumulate.
// One block (8 waves) per (b, pair). Wave w = output column tile; loops all
// 8 row tiles. K=256 -> 8 steps of 16x16x32 WMMA. 64 WMMAs per wave.
// ---------------------------------------------------------------------------
__global__ void __launch_bounds__(256)
hessian_kernel(const float* __restrict__ cbuf,
               const _Float16* __restrict__ swz,
               float* __restrict__ out) {
    __shared__ float c_sh[HID];
    int b    = blockIdx.x;
    int pair = blockIdx.y;
    int i    = pair / 3;
    int jr   = pair % 3;
    int j    = jr + (jr >= i ? 1 : 0);

    c_sh[threadIdx.x] = cbuf[((size_t)(pair * BATCH + b)) * HID + threadIdx.x];
    __syncthreads();

    int w    = threadIdx.x >> 5;   // wave id == n-tile
    int lane = threadIdx.x & 31;
    int half = lane >> 4;

    v8f acc[8];
#pragma unroll
    for (int mt = 0; mt < 8; ++mt) acc[mt] = {};

    const uint4* sbase = (const uint4*)(swz + (size_t)i * (8 * 8 * 32 * 16));

#pragma unroll
    for (int kb = 0; kb < 8; ++kb) {
        // raw W1 fragment for column tile w, then scale by c -> B fragment
        Frag16 braw;
        const uint4* bp = sbase + ((size_t)((kb * 8 + w) * 32 + lane)) * 2;
        braw.q[0] = bp[0];
        braw.q[1] = bp[1];
        Frag16 bs;
#pragma unroll
        for (int e = 0; e < 16; ++e) {
            int kp  = (e & 7) + ((e >> 3) << 4) + (half << 3);
            float cv = c_sh[kb * 32 + kp];           // broadcast per half-wave
            bs.h[e] = (_Float16)((float)braw.h[e] * cv);
        }
#pragma unroll
        for (int mt = 0; mt < 8; ++mt) {
            Frag16 af;  // A = W1^T fragment for row tile mt (same buffer)
            const uint4* ap = sbase + ((size_t)((kb * 8 + mt) * 32 + lane)) * 2;
            af.q[0] = ap[0];
            af.q[1] = ap[1];
            acc[mt] = __builtin_amdgcn_wmma_f32_16x16x32_f16(
                false, af.v, false, bs.v, (short)0, acc[mt], false, false);
        }
    }

    // C/D layout: VGPR r, lanes 0-15 -> M=r, N=lane; lanes 16-31 -> M=8+r.
    float* outp = out + ((((size_t)i * NAG + j) * BATCH + b) << 14);  // *128*128
    int col = w * 16 + (lane & 15);
#pragma unroll
    for (int mt = 0; mt < 8; ++mt) {
#pragma unroll
        for (int r = 0; r < 8; ++r) {
            int row = mt * 16 + r + half * 8;
            outp[(size_t)row * DLAT + col] = acc[mt][r];
        }
    }
}

// ---------------------------------------------------------------------------
extern "C" void kernel_launch(void* const* d_in, const int* in_sizes, int n_in,
                              void* d_out, int out_size, void* d_ws, size_t ws_size,
                              hipStream_t stream) {
    (void)in_sizes; (void)n_in; (void)out_size; (void)ws_size;
    const float* z_all = (const float*)d_in[0];
    const float* W1    = (const float*)d_in[1];
    const float* b1    = (const float*)d_in[2];
    const float* W2    = (const float*)d_in[3];
    // d_in[4] = b2: drops out of the Hessian.
    float* out = (float*)d_out;

    // workspace: [0, 256KB) swizzled f16 W1 fragments; [256KB, 256KB+3MB) coeffs
    _Float16* swz  = (_Float16*)d_ws;
    float*    cbuf = (float*)((char*)d_ws + (size_t)NAG * 8 * 8 * 32 * 16 * 2);

    swizzle_w1_kernel<<<32, 256, 0, stream>>>(W1, swz);
    coeff_kernel<<<dim3(BATCH, NPAIR), 256, 0, stream>>>(z_all, W1, b1, W2, cbuf);
    zero_diag_kernel<<<2048, 256, 0, stream>>>(out);
    hessian_kernel<<<dim3(BATCH, NPAIR), 256, 0, stream>>>(cbuf, swz, out);
}
